// MultiHaedCrossAttention_75144747811345
// MI455X (gfx1250) — matrix operand order repaired and checked by reading, not compiled, not run on hardware
//
#include <hip/hip_runtime.h>

// ---------------------------------------------------------------------------
// MI455X (gfx1250) multi-head cross attention, CDNA5-native data path:
//   f32 -> bf16 once (weights transposed), TDM (tensor_load_to_lds) -> LDS,
//   every WMMA operand is a contiguous 32B LDS vector load, v_wmma bf16,
//   flash-style fused softmax, fp32 accumulate.
// ---------------------------------------------------------------------------

typedef __attribute__((ext_vector_type(16))) __bf16       bf16x16;
typedef __attribute__((ext_vector_type(8)))  __bf16       bf16x8;
typedef __attribute__((ext_vector_type(4)))  __bf16       bf16x4;
typedef __attribute__((ext_vector_type(8)))  float        f32x8;
typedef __attribute__((ext_vector_type(4)))  float        f32x4;
typedef __attribute__((ext_vector_type(4)))  unsigned int u32x4;
typedef __attribute__((ext_vector_type(4)))  int          i32x4;
typedef __attribute__((ext_vector_type(8)))  int          i32x8;

#define DIM_C 1024

__device__ __forceinline__ f32x8 wmma_bf16(bf16x16 a, bf16x16 b, f32x8 c) {
  return __builtin_amdgcn_wmma_f32_16x16x32_bf16(
      false, a, false, b, (short)0, c, false, false);
}

__device__ __forceinline__ bf16x16 cat8(bf16x8 lo, bf16x8 hi) {
  return __builtin_shufflevector(lo, hi, 0, 1, 2, 3, 4, 5, 6, 7,
                                         8, 9, 10, 11, 12, 13, 14, 15);
}

__device__ __forceinline__ float redmax16(float x) {
#pragma unroll
  for (int m = 1; m < 16; m <<= 1) x = fmaxf(x, __shfl_xor(x, m, 32));
  return x;
}
__device__ __forceinline__ float redsum16(float x) {
#pragma unroll
  for (int m = 1; m < 16; m <<= 1) x += __shfl_xor(x, m, 32);
  return x;
}

// ---------------------------------------------------------------------------
// TDM: async 2-D bf16 tile load, global -> LDS (descriptor per ISA ch.8).
// Tracked by TENSORcnt; caller pairs with s_wait_tensorcnt + barrier.
// ---------------------------------------------------------------------------
__device__ __forceinline__ void tdm_load_2d_bf16(unsigned lds_off,
                                                 const __bf16* gsrc,
                                                 unsigned tile_w, unsigned tile_h,
                                                 unsigned stride_elems) {
  unsigned long long ga = (unsigned long long)(size_t)gsrc;
  u32x4 g0;
  g0[0] = 1u;                                         // count=1 (valid), user mode
  g0[1] = lds_off;                                    // LDS byte address
  g0[2] = (unsigned)ga;                               // global_addr[31:0]
  g0[3] = (unsigned)((ga >> 32) & 0x01FFFFFFu) | (2u << 30);  // [56:32], type=2
  i32x8 g1;
  g1[0] = (int)(1u << 16);                            // wg_mask=0, data_size=1 (2B)
  g1[1] = (int)((tile_w & 0xFFFFu) << 16);            // tensor_dim0[15:0]
  g1[2] = (int)((tile_w >> 16) | ((tile_h & 0xFFFFu) << 16)); // td0 hi | td1 lo
  g1[3] = (int)((tile_h >> 16) | ((tile_w & 0xFFFFu) << 16)); // td1 hi | tile_dim0
  g1[4] = (int)(tile_h & 0xFFFFu);                    // tile_dim1, tile_dim2=0
  g1[5] = (int)stride_elems;                          // tensor_dim0_stride[31:0]
  g1[6] = 0;
  g1[7] = 0;
  i32x4 z4 = {0, 0, 0, 0};
#if defined(__clang_major__) && (__clang_major__ >= 23)
  i32x8 z8 = {0, 0, 0, 0, 0, 0, 0, 0};
  __builtin_amdgcn_tensor_load_to_lds(g0, g1, z4, z4, z8, 0);
#else
  __builtin_amdgcn_tensor_load_to_lds(g0, g1, z4, z4, 0);
#endif
}

// ---------------------------------------------------------------------------
// Elementwise f32 -> bf16 (vectorized x4).
// ---------------------------------------------------------------------------
__global__ void f32_to_bf16(const float* __restrict__ in,
                            __bf16* __restrict__ out, int n4) {
  int i = blockIdx.x * blockDim.x + threadIdx.x;
  if (i >= n4) return;
  f32x4 v = ((const f32x4*)in)[i];
  bf16x4 o;
  o[0] = (__bf16)v[0]; o[1] = (__bf16)v[1];
  o[2] = (__bf16)v[2]; o[3] = (__bf16)v[3];
  ((bf16x4*)out)[i] = o;
}

// ---------------------------------------------------------------------------
// f32 [R][C] -> bf16 transposed [C][R], 16x16 LDS tiles (both sides coalesced).
// ---------------------------------------------------------------------------
__global__ void f32_to_bf16_T(const float* __restrict__ in,
                              __bf16* __restrict__ out, int R, int C) {
  __shared__ float t[16][17];
  int r = blockIdx.y * 16 + threadIdx.y;
  int c = blockIdx.x * 16 + threadIdx.x;
  t[threadIdx.y][threadIdx.x] = in[(size_t)r * C + c];
  __syncthreads();
  int ro = blockIdx.x * 16 + threadIdx.y;   // output row = input col
  int co = blockIdx.y * 16 + threadIdx.x;   // output col = input row
  out[(size_t)ro * R + co] = (__bf16)t[threadIdx.x][threadIdx.y];
}

// ---------------------------------------------------------------------------
// C[M,N] = (A_bf16[M,K] @ B, B given TRANSPOSED as BwT[N,K]) * scale (+bias).
// TDM double-buffered; every WMMA operand is a contiguous LDS vector load.
// grid = (N/64, M/64), block = 128 (4 waves). Outputs (one of):
//   Cb  : bf16 row-major [M][N]
//   CbT : bf16 batch-transposed [M/TN][N][TN]   (used for V^T)
//   Cf  : f32 row-major + bias
// ---------------------------------------------------------------------------
__global__ void gemm_bf16_tdm(const __bf16* __restrict__ A,
                              const __bf16* __restrict__ BwT,
                              const float* __restrict__ bias,
                              float* __restrict__ Cf,
                              __bf16* __restrict__ Cb,
                              __bf16* __restrict__ CbT,
                              int M, int K, int N, int TN, float scale) {
  __shared__ __bf16 As [2][64][32];   // [buf][m][k]
  __shared__ __bf16 BsT[2][64][32];   // [buf][n][k]  (transposed weight tile)

  const int lane = threadIdx.x & 31;
  const int wave = threadIdx.x >> 5;
  const int g    = lane >> 4;
  const int ln   = lane & 15;
  const int kbA  = g * 16;
  const int kbB  = g * 16;
  const int tileN = blockIdx.x * 64;
  const int tileM = blockIdx.y * 64;

  f32x8 acc[4];
#pragma unroll
  for (int t = 0; t < 4; ++t)
#pragma unroll
    for (int i = 0; i < 8; ++i) acc[t][i] = 0.0f;

  const __bf16* Abase = A   + (size_t)tileM * K;
  const __bf16* Bbase = BwT + (size_t)tileN * K;

  if (wave == 0) {
    tdm_load_2d_bf16((unsigned)(size_t)&As [0][0][0], Abase, 32, 64, (unsigned)K);
    tdm_load_2d_bf16((unsigned)(size_t)&BsT[0][0][0], Bbase, 32, 64, (unsigned)K);
    __builtin_amdgcn_s_wait_tensorcnt(0);
  }
  __syncthreads();

  int buf = 0;
  for (int k0 = 0; k0 < K; k0 += 32) {
    if (wave == 0 && (k0 + 32) < K) {   // prefetch next chunk into other buffer
      tdm_load_2d_bf16((unsigned)(size_t)&As [buf ^ 1][0][0],
                       Abase + (k0 + 32), 32, 64, (unsigned)K);
      tdm_load_2d_bf16((unsigned)(size_t)&BsT[buf ^ 1][0][0],
                       Bbase + (k0 + 32), 32, 64, (unsigned)K);
    }

    // A fragment: two contiguous 16B runs
    bf16x16 a = cat8(*(const bf16x8*)&As[buf][wave * 16 + ln][kbA],
                     *(const bf16x8*)&As[buf][wave * 16 + ln][kbA + 8]);
#pragma unroll
    for (int t = 0; t < 4; ++t) {
      // B fragment: ONE contiguous 32B LDS load (lane=n, elements=k)
      bf16x16 b = *(const bf16x16*)&BsT[buf][t * 16 + ln][kbB];
      acc[t] = wmma_bf16(a, b, acc[t]);
    }

    if (wave == 0) __builtin_amdgcn_s_wait_tensorcnt(0);
    __syncthreads();
    buf ^= 1;
  }

  const int rbase = tileM + wave * 16 + g * 8;
  if (Cb) {
#pragma unroll
    for (int t = 0; t < 4; ++t)
#pragma unroll
      for (int i = 0; i < 8; ++i)
        Cb[(size_t)(rbase + i) * N + tileN + t * 16 + ln] =
            (__bf16)(acc[t][i] * scale);
  } else if (CbT) {
    // batch-transposed store: out[b][col][row_in_batch]  (tiles never cross b)
    const int bidx = tileM / TN;
    const int r0   = rbase - bidx * TN;
    __bf16* basep = CbT + (size_t)bidx * N * TN;
#pragma unroll
    for (int t = 0; t < 4; ++t)
#pragma unroll
      for (int i = 0; i < 8; ++i)
        basep[(size_t)(tileN + t * 16 + ln) * TN + r0 + i] =
            (__bf16)(acc[t][i] * scale);
  } else {
#pragma unroll
    for (int t = 0; t < 4; ++t) {
      const float bb = bias ? bias[tileN + t * 16 + ln] : 0.0f;
#pragma unroll
      for (int i = 0; i < 8; ++i)
        Cf[(size_t)(rbase + i) * N + tileN + t * 16 + ln] = acc[t][i] * scale + bb;
    }
  }
}

// ---------------------------------------------------------------------------
// Fused flash attention, TDM-fed. Q pre-scaled by DIM^-0.5.
// K is row-major [nkv][DIM]; V is pre-transposed per batch: Vt[b][DIM][NKV].
// grid = (B*H, NQ/64), block = 128 (4 waves, 16 query rows each).
// ---------------------------------------------------------------------------
__global__ void attn_fused_tdm(const __bf16* __restrict__ Q,
                               const __bf16* __restrict__ Km,
                               const __bf16* __restrict__ Vt,
                               __bf16* __restrict__ Z,
                               int NQ, int NKV) {
  __shared__ __bf16 Ks [2][32][64];   // [buf][kv row][head col]
  __shared__ __bf16 VsT[2][64][32];   // [buf][head col][kv row]
  __shared__ __bf16 plds[4][16][32];  // per-wave P transpose staging

  const int lane = threadIdx.x & 31;
  const int wave = threadIdx.x >> 5;
  const int g    = lane >> 4;
  const int ln   = lane & 15;
  const int kbA  = g * 16;
  const int kbB  = g * 16;

  const int b  = blockIdx.x >> 4;   // H = 16
  const int h  = blockIdx.x & 15;
  const int hd = h * 64;
  const int qt = blockIdx.y * 64 + wave * 16;

  const __bf16* Qb  = Q  + ((size_t)(b * NQ + qt)) * DIM_C + hd;
  const __bf16* Kb  = Km + (size_t)b * NKV * DIM_C + hd;
  const __bf16* VbT = Vt + (size_t)b * DIM_C * NKV + (size_t)hd * NKV;

  if (wave == 0) {
    tdm_load_2d_bf16((unsigned)(size_t)&Ks [0][0][0], Kb,  64, 32, DIM_C);
    tdm_load_2d_bf16((unsigned)(size_t)&VsT[0][0][0], VbT, 32, 64, (unsigned)NKV);
    __builtin_amdgcn_s_wait_tensorcnt(0);
  }

  // Q fragments (head dim 64 -> k chunks 0..31, 32..63)
  bf16x16 q0, q1;
  {
    const __bf16* qp = Qb + (size_t)ln * DIM_C;
    q0 = cat8(*(const bf16x8*)&qp[kbA],      *(const bf16x8*)&qp[8 + kbA]);
    q1 = cat8(*(const bf16x8*)&qp[32 + kbA], *(const bf16x8*)&qp[40 + kbA]);
  }

  float mrun[8], srun[8];
  f32x8 zacc[4];
#pragma unroll
  for (int i = 0; i < 8; ++i) { mrun[i] = -3.0e38f; srun[i] = 0.0f; }
#pragma unroll
  for (int t = 0; t < 4; ++t)
#pragma unroll
    for (int i = 0; i < 8; ++i) zacc[t][i] = 0.0f;

  __syncthreads();

  int buf = 0;
  for (int j0 = 0; j0 < NKV; j0 += 32) {
    if (wave == 0 && (j0 + 32) < NKV) {  // prefetch next K/V chunk
      tdm_load_2d_bf16((unsigned)(size_t)&Ks [buf ^ 1][0][0],
                       Kb + (size_t)(j0 + 32) * DIM_C, 64, 32, DIM_C);
      tdm_load_2d_bf16((unsigned)(size_t)&VsT[buf ^ 1][0][0],
                       VbT + (j0 + 32), 32, 64, (unsigned)NKV);
    }

    // ---- S = Q @ K^T: contiguous 32B K-fragment loads ----
    f32x8 s0, s1;
#pragma unroll
    for (int i = 0; i < 8; ++i) { s0[i] = 0.0f; s1[i] = 0.0f; }
    {
      bf16x16 bk;
      bk = *(const bf16x16*)&Ks[buf][ln][kbB];           s0 = wmma_bf16(q0, bk, s0);
      bk = *(const bf16x16*)&Ks[buf][ln][32 + kbB];      s0 = wmma_bf16(q1, bk, s0);
      bk = *(const bf16x16*)&Ks[buf][16 + ln][kbB];      s1 = wmma_bf16(q0, bk, s1);
      bk = *(const bf16x16*)&Ks[buf][16 + ln][32 + kbB]; s1 = wmma_bf16(q1, bk, s1);
    }

    // ---- online softmax over this 32-column slab ----
#pragma unroll
    for (int i = 0; i < 8; ++i) {
      float mloc = redmax16(fmaxf(s0[i], s1[i]));
      float mn   = fmaxf(mrun[i], mloc);
      float corr = __expf(mrun[i] - mn);
      mrun[i] = mn;
      float p0 = __expf(s0[i] - mn);
      float p1 = __expf(s1[i] - mn);
      srun[i] = srun[i] * corr + redsum16(p0 + p1);
#pragma unroll
      for (int t = 0; t < 4; ++t) zacc[t][i] *= corr;
      plds[wave][g * 8 + i][ln]      = (__bf16)p0;   // C-layout -> LDS
      plds[wave][g * 8 + i][16 + ln] = (__bf16)p1;
    }
    __syncthreads();

    // ---- reload P as A-fragment ----
    bf16x16 pa = cat8(*(const bf16x8*)&plds[wave][ln][kbA],
                      *(const bf16x8*)&plds[wave][ln][8 + kbA]);

    // ---- Z += P @ V_chunk: contiguous 32B V^T-fragment loads ----
#pragma unroll
    for (int t = 0; t < 4; ++t) {
      bf16x16 bv = *(const bf16x16*)&VsT[buf][t * 16 + ln][kbB];
      zacc[t] = wmma_bf16(pa, bv, zacc[t]);
    }

    if (wave == 0) __builtin_amdgcn_s_wait_tensorcnt(0);
    __syncthreads();
    buf ^= 1;
  }

  // ---- epilogue ----
#pragma unroll
  for (int t = 0; t < 4; ++t)
#pragma unroll
    for (int i = 0; i < 8; ++i) {
      float z = zacc[t][i] / srun[i];
      Z[(size_t)(b * NQ + qt + g * 8 + i) * DIM_C + hd + t * 16 + ln] = (__bf16)z;
    }
}

// ---------------------------------------------------------------------------
// Host launcher. Inputs: x1,x2,x3,Wq,Wk,Wv,Wo,bo (f32). Output f32.
// Workspace (bf16): converted x's, transposed weights, Q, K, V^T, Z (~84 MB).
// ---------------------------------------------------------------------------
extern "C" void kernel_launch(void* const* d_in, const int* in_sizes, int n_in,
                              void* d_out, int out_size, void* d_ws, size_t ws_size,
                              hipStream_t stream) {
  const float* x1 = (const float*)d_in[0];
  const float* x2 = (const float*)d_in[1];
  const float* x3 = (const float*)d_in[2];
  const float* Wq = (const float*)d_in[3];
  const float* Wk = (const float*)d_in[4];
  const float* Wv = (const float*)d_in[5];
  const float* Wo = (const float*)d_in[6];
  const float* bo = (const float*)d_in[7];
  float* out = (float*)d_out;

  const int B = 4, NQ = 1024, NKV = 1536, DIM = 1024;
  const int MQ = B * NQ;    // 4096
  const int MK = B * NKV;   // 6144

  __bf16* p    = (__bf16*)d_ws;
  __bf16* cx1  = p; p += (size_t)MQ * DIM;
  __bf16* cx2  = p; p += (size_t)MK * DIM;
  __bf16* cx3  = p; p += (size_t)MK * DIM;
  __bf16* WqT  = p; p += (size_t)DIM * DIM;
  __bf16* WkT  = p; p += (size_t)DIM * DIM;
  __bf16* WvT  = p; p += (size_t)DIM * DIM;
  __bf16* WoT  = p; p += (size_t)DIM * DIM;
  __bf16* Qb   = p; p += (size_t)MQ * DIM;
  __bf16* Kb   = p; p += (size_t)MK * DIM;
  __bf16* VtB  = p; p += (size_t)MK * DIM;   // per-batch transposed V
  __bf16* Zb   = p;

  auto conv = [&](const float* src, __bf16* dst, size_t n) {
    int n4 = (int)(n / 4);
    f32_to_bf16<<<(n4 + 255) / 256, 256, 0, stream>>>(src, dst, n4);
  };
  conv(x1, cx1, (size_t)MQ * DIM);
  conv(x2, cx2, (size_t)MK * DIM);
  conv(x3, cx3, (size_t)MK * DIM);

  dim3 tg(DIM / 16, DIM / 16), tb(16, 16);
  f32_to_bf16_T<<<tg, tb, 0, stream>>>(Wq, WqT, DIM, DIM);
  f32_to_bf16_T<<<tg, tb, 0, stream>>>(Wk, WkT, DIM, DIM);
  f32_to_bf16_T<<<tg, tb, 0, stream>>>(Wv, WvT, DIM, DIM);
  f32_to_bf16_T<<<tg, tb, 0, stream>>>(Wo, WoT, DIM, DIM);

  // 1) projections (DIM^-0.5 folded into Q); V written batch-transposed
  gemm_bf16_tdm<<<dim3(DIM / 64, MQ / 64), 128, 0, stream>>>(
      cx1, WqT, nullptr, nullptr, Qb, nullptr, MQ, DIM, DIM, 0, 0.03125f);
  gemm_bf16_tdm<<<dim3(DIM / 64, MK / 64), 128, 0, stream>>>(
      cx2, WkT, nullptr, nullptr, Kb, nullptr, MK, DIM, DIM, 0, 1.0f);
  gemm_bf16_tdm<<<dim3(DIM / 64, MK / 64), 128, 0, stream>>>(
      cx3, WvT, nullptr, nullptr, nullptr, VtB, MK, DIM, DIM, NKV, 1.0f);

  // 2) fused flash attention
  attn_fused_tdm<<<dim3(B * 16, NQ / 64), 128, 0, stream>>>(Qb, Kb, VtB, Zb, NQ, NKV);

  // 3) output projection + bias (f32 out)
  gemm_bf16_tdm<<<dim3(DIM / 64, MQ / 64), 128, 0, stream>>>(
      Zb, WoT, bo, out, nullptr, nullptr, MQ, DIM, DIM, 0, 1.0f);
}
